// Attention_7962869366891
// MI455X (gfx1250) — matrix operand order repaired and checked by reading, not compiled, not run on hardware
//
#include <hip/hip_runtime.h>
#include <hip/hip_bf16.h>

// Shapes (fixed by the reference): B=4, N=2048, F=1024, H=16, D=64, HD=1024
// Rows R = B*N = 8192. QKV combined col dim E = 3072.

typedef __attribute__((ext_vector_type(16))) _Float16 v16h;
typedef __attribute__((ext_vector_type(8)))  _Float16 v8h;
typedef __attribute__((ext_vector_type(8)))  float    v8f;
typedef int i4vec __attribute__((vector_size(16)));   // matches builtin's V4i

static __device__ __forceinline__ v16h cat8(v8h lo, v8h hi) {
  return __builtin_shufflevector(lo, hi, 0,1,2,3,4,5,6,7,8,9,10,11,12,13,14,15);
}
static __device__ __forceinline__ v8f wmma16(v16h a, v16h b, v8f c) {
  // D = A(16x32 f16) * B(32x16 f16) + C(16x16 f32)
  return __builtin_amdgcn_wmma_f32_16x16x32_f16(false, a, false, b, (short)0, c, false, false);
}
static __device__ __forceinline__ float rmax16(float v) {
  v = fmaxf(v, __shfl_xor(v, 1, 32));
  v = fmaxf(v, __shfl_xor(v, 2, 32));
  v = fmaxf(v, __shfl_xor(v, 4, 32));
  v = fmaxf(v, __shfl_xor(v, 8, 32));
  return v;
}
static __device__ __forceinline__ float rsum16(float v) {
  v += __shfl_xor(v, 1, 32);
  v += __shfl_xor(v, 2, 32);
  v += __shfl_xor(v, 4, 32);
  v += __shfl_xor(v, 8, 32);
  return v;
}

// ---- async copy global -> LDS (16B per lane), tracked by ASYNCcnt ----------
static __device__ __forceinline__ void async_cp16(const _Float16* g, _Float16* l) {
#if defined(__HIP_DEVICE_COMPILE__) && __has_builtin(__builtin_amdgcn_global_load_async_to_lds_b128)
  __builtin_amdgcn_global_load_async_to_lds_b128(
      (__attribute__((address_space(1))) i4vec*)g,
      (__attribute__((address_space(3))) i4vec*)l, 0, 0);
#else
  *(v8h*)l = *(const v8h*)g;   // synchronous fallback
#endif
}
#if defined(__HIP_DEVICE_COMPILE__) && __has_builtin(__builtin_amdgcn_s_wait_asynccnt)
#define WAIT_ASYNC(n) __builtin_amdgcn_s_wait_asynccnt(n)
#elif defined(__HIP_DEVICE_COMPILE__) && __has_builtin(__builtin_amdgcn_global_load_async_to_lds_b128)
#define WAIT_ASYNC(n) asm volatile("s_wait_asynccnt %0" ::"i"(n) : "memory")
#else
#define WAIT_ASYNC(n) do {} while (0)
#endif

// ---------------- Kernel 1: fused dual layernorm + f16 cast -----------------
__global__ void __launch_bounds__(256) k_ln2_cast(
    const float* __restrict__ x,
    const float* __restrict__ g0, const float* __restrict__ b0,
    const float* __restrict__ g1, const float* __restrict__ b1,
    _Float16* __restrict__ xn, _Float16* __restrict__ cn) {
  const int row = blockIdx.x;
  const int tid = threadIdx.x;
  const float* xr = x + (size_t)row * 1024;
  float xv[4];
  float s = 0.f, q = 0.f;
#pragma unroll
  for (int i = 0; i < 4; ++i) {
    xv[i] = xr[tid + i * 256];
    s += xv[i];
    q += xv[i] * xv[i];
  }
#pragma unroll
  for (int m = 16; m >= 1; m >>= 1) {
    s += __shfl_xor(s, m, 32);
    q += __shfl_xor(q, m, 32);
  }
  __shared__ float ss[8], sq[8];
  if ((tid & 31) == 0) { ss[tid >> 5] = s; sq[tid >> 5] = q; }
  __syncthreads();
  float ts = 0.f, tq = 0.f;
#pragma unroll
  for (int i = 0; i < 8; ++i) { ts += ss[i]; tq += sq[i]; }
  const float mu   = ts * (1.f / 1024.f);
  const float var  = tq * (1.f / 1024.f) - mu * mu;
  const float rstd = rsqrtf(var + 1e-5f);
#pragma unroll
  for (int i = 0; i < 4; ++i) {
    const int c = tid + i * 256;
    const float xc = (xv[i] - mu) * rstd;
    xn[(size_t)row * 1024 + c] = (_Float16)(xc * g0[c] + b0[c]);
    cn[(size_t)row * 1024 + c] = (_Float16)(xc * g1[c] + b1[c]);
  }
}

// ---------------- Kernel 2: f32 -> f16 weight cast --------------------------
__global__ void k_cvt_f16(const float* __restrict__ s, _Float16* __restrict__ d, int n) {
  for (int i = blockIdx.x * blockDim.x + threadIdx.x; i < n; i += gridDim.x * blockDim.x)
    d[i] = (_Float16)s[i];
}

// ---------------- Kernel 3: QKV projection (WMMA + async LDS weights) -------
// Block = 8 waves = 128 rows x 64 cols. The 64x32 weight tile per k-step is
// staged in LDS once per block (double buffered, ASYNCcnt) and shared by all
// 8 waves; A fragments stream directly from global (per-wave-unique rows).
__global__ void __launch_bounds__(256) k_gemm_qkv(
    const _Float16* __restrict__ xn, const _Float16* __restrict__ cn,
    const _Float16* __restrict__ wq, const _Float16* __restrict__ wkv,
    _Float16* __restrict__ Qh, _Float16* __restrict__ Kh, _Float16* __restrict__ Vt) {
  __shared__ __align__(16) _Float16 wtile[2][64][32];   // [buf][e_local][k]
  const int tid = threadIdx.x;
  const int w = tid >> 5, lane = tid & 31;
  const int half = lane >> 4, r = lane & 15;
  const int ct = blockIdx.x % 48;        // 48 col jobs (E=3072 / 64)
  const int rg = blockIdx.x / 48;        // 64 row groups (8192 / 128)
  const int e0 = ct * 64;
  const int n0 = rg * 128 + w * 16;

  const _Float16* A = (e0 < 1024) ? xn : cn;
  const _Float16* arow = A + (size_t)(n0 + r) * 1024;

  // staging map: thread -> (weight row e0 + tid/4, 8-half chunk (tid%4)*8)
  const int fe = e0 + (tid >> 2);
  const _Float16* fsrc = (fe < 1024) ? (wq + (size_t)fe * 1024)
                                     : (wkv + (size_t)(fe - 1024) * 1024);
  const int fk = (tid & 3) * 8;
  _Float16* fdst[2] = { &wtile[0][tid >> 2][fk], &wtile[1][tid >> 2][fk] };

  const v8f zero = {};
  v8f acc[4];
#pragma unroll
  for (int t = 0; t < 4; ++t) acc[t] = zero;

  async_cp16(fsrc + fk, fdst[0]);
  for (int k0 = 0; k0 < 1024; k0 += 32) {
    const int buf = (k0 >> 5) & 1;
    const bool more = (k0 + 32) < 1024;
    if (more) {
      async_cp16(fsrc + k0 + 32 + fk, fdst[buf ^ 1]);
      WAIT_ASYNC(1);
    } else {
      WAIT_ASYNC(0);
    }
    __syncthreads();
    const v16h a = cat8(*(const v8h*)(arow + k0 + half * 16),
                        *(const v8h*)(arow + k0 + half * 16 + 8));
#pragma unroll
    for (int t = 0; t < 4; ++t) {
      const v16h bf = *(const v16h*)&wtile[buf][t * 16 + r][half * 16];
      acc[t] = wmma16(a, bf, acc[t]);
    }
    __syncthreads();
  }

  const int b = n0 >> 11, n_in = n0 & 2047;
  if (e0 < 2048) {                      // Q or K: row-major [b,h,n,d]
    _Float16* dst; int h;
    if (e0 < 1024) { h = e0 >> 6; dst = Qh; }
    else           { h = (e0 - 1024) >> 6; dst = Kh; }
    const size_t base = ((size_t)(b * 16 + h) * 2048 + n_in) * 64;
#pragma unroll
    for (int t = 0; t < 4; ++t)
#pragma unroll
      for (int v = 0; v < 8; ++v)
        dst[base + (size_t)(v + 8 * half) * 64 + t * 16 + r] = (_Float16)acc[t][v];
  } else {                              // V transposed: [b,h,d,n]; 16B store/lane
    const int h = (e0 - 2048) >> 6;
#pragma unroll
    for (int t = 0; t < 4; ++t) {
      v8h pk;
#pragma unroll
      for (int v = 0; v < 8; ++v) pk[v] = (_Float16)acc[t][v];
      const size_t base =
          ((size_t)(b * 16 + h) * 64 + (t * 16 + r)) * 2048 + n_in + 8 * half;
      *(v8h*)(Vt + base) = pk;
    }
  }
}

// ---------------- Kernel 4: flash attention (WMMA + async LDS K/V) ----------
// All 8 waves of a block share one (b,h); K (32x64) and V^T (64x32) chunks are
// staged in LDS once per block per 32-key step, double buffered on ASYNCcnt.
__global__ void __launch_bounds__(256) k_attn(
    const _Float16* __restrict__ Qh, const _Float16* __restrict__ Kh,
    const _Float16* __restrict__ Vt, _Float16* __restrict__ Oh) {
  __shared__ __align__(16) _Float16 ktile[2][32][64];   // [buf][key][d]
  __shared__ __align__(16) _Float16 vtile[2][64][32];   // [buf][d][key]
  __shared__ __align__(16) _Float16 pbuf[8][16][32];    // per-wave P re-layout
  const int tid = threadIdx.x;
  const int w = tid >> 5, lane = tid & 31;
  const int half = lane >> 4, r = lane & 15;
  const int bh = blockIdx.x >> 4;                       // 64 (b,h)
  const int n0 = ((blockIdx.x & 15) * 8 + w) * 16;      // q tile row

  const _Float16* qrow = Qh + ((size_t)bh * 2048 + n0 + r) * 64;
  const v16h qa0 = cat8(*(const v8h*)(qrow + half * 16),
                        *(const v8h*)(qrow + half * 16 + 8));
  const v16h qa1 = cat8(*(const v8h*)(qrow + 32 + half * 16),
                        *(const v8h*)(qrow + 40 + half * 16));

  const v8f zero = {};
  v8f acc[4];
#pragma unroll
  for (int t = 0; t < 4; ++t) acc[t] = zero;
  float mrow[8], lrow[8];
#pragma unroll
  for (int v = 0; v < 8; ++v) { mrow[v] = -1e30f; lrow[v] = 0.f; }

  const _Float16* kbase = Kh + (size_t)bh * 2048 * 64;
  const _Float16* vbase = Vt + (size_t)bh * 64 * 2048;

  // staging maps (16B per thread per tile)
  const int kkey = tid >> 3, kchunk = (tid & 7) * 8;    // ktile: 32 rows x 128B
  const int vd   = tid >> 2, vchunk = (tid & 3) * 8;    // vtile: 64 rows x 64B
  const _Float16* ksrc = kbase + (size_t)kkey * 64 + kchunk;
  const _Float16* vsrc = vbase + (size_t)vd * 2048 + vchunk;

  async_cp16(ksrc, &ktile[0][kkey][kchunk]);
  async_cp16(vsrc, &vtile[0][vd][vchunk]);
  for (int m0 = 0; m0 < 2048; m0 += 32) {
    const int buf = (m0 >> 5) & 1;
    const bool more = (m0 + 32) < 2048;
    if (more) {
      async_cp16(ksrc + (size_t)(m0 + 32) * 64, &ktile[buf ^ 1][kkey][kchunk]);
      async_cp16(vsrc + m0 + 32,                &vtile[buf ^ 1][vd][vchunk]);
      WAIT_ASYNC(2);
    } else {
      WAIT_ASYNC(0);
    }
    __syncthreads();

    v8f s0 = wmma16(qa0, *(const v16h*)&ktile[buf][r][half * 16], zero);
    s0     = wmma16(qa1, *(const v16h*)&ktile[buf][r][32 + half * 16], s0);
    v8f s1 = wmma16(qa0, *(const v16h*)&ktile[buf][16 + r][half * 16], zero);
    s1     = wmma16(qa1, *(const v16h*)&ktile[buf][16 + r][32 + half * 16], s1);

#pragma unroll
    for (int v = 0; v < 8; ++v) {
      const float a0 = s0[v] * 0.125f;   // D^-0.5
      const float a1 = s1[v] * 0.125f;
      const float mx   = rmax16(fmaxf(a0, a1));
      const float mnew = fmaxf(mrow[v], mx);
      const float alpha = __expf(mrow[v] - mnew);
      const float p0 = __expf(a0 - mnew);
      const float p1 = __expf(a1 - mnew);
      lrow[v] = lrow[v] * alpha + rsum16(p0 + p1);
      mrow[v] = mnew;
#pragma unroll
      for (int t = 0; t < 4; ++t) acc[t][v] *= alpha;
      pbuf[w][v + 8 * half][r]      = (_Float16)p0;
      pbuf[w][v + 8 * half][16 + r] = (_Float16)p1;
    }
    // wave-local re-layout: (row in VGPR, key in lane) -> A-frag layout
    const v16h pa = cat8(*(const v8h*)&pbuf[w][r][half * 16],
                         *(const v8h*)&pbuf[w][r][half * 16 + 8]);
#pragma unroll
    for (int t = 0; t < 4; ++t)
      acc[t] = wmma16(pa, *(const v16h*)&vtile[buf][t * 16 + r][half * 16], acc[t]);
    __syncthreads();
  }

  const int b = bh >> 4, h = bh & 15;
  const size_t obase = ((size_t)b * 2048 + n0) * 1024 + h * 64;
#pragma unroll
  for (int v = 0; v < 8; ++v) {
    const float inv = 1.f / lrow[v];
#pragma unroll
    for (int t = 0; t < 4; ++t)
      Oh[obase + (size_t)(v + 8 * half) * 1024 + t * 16 + r] =
          (_Float16)(acc[t][v] * inv);
  }
}

// ---------------- Kernel 5: output projection + residual (WMMA) -------------
__global__ void __launch_bounds__(256) k_gemm_out(
    const _Float16* __restrict__ Oh, const _Float16* __restrict__ wo,
    const float* __restrict__ x, float* __restrict__ out) {
  __shared__ __align__(16) _Float16 wtile[2][64][32];
  const int tid = threadIdx.x;
  const int w = tid >> 5, lane = tid & 31;
  const int half = lane >> 4, r = lane & 15;
  const int ct = blockIdx.x & 15;        // 16 col jobs (F=1024 / 64)
  const int rg = blockIdx.x >> 4;        // 64 row groups
  const int f0 = ct * 64;
  const int n0 = rg * 128 + w * 16;

  const _Float16* arow = Oh + (size_t)(n0 + r) * 1024;
  const _Float16* fsrc = wo + (size_t)(f0 + (tid >> 2)) * 1024;
  const int fk = (tid & 3) * 8;
  _Float16* fdst[2] = { &wtile[0][tid >> 2][fk], &wtile[1][tid >> 2][fk] };

  const v8f zero = {};
  v8f acc[4];
#pragma unroll
  for (int t = 0; t < 4; ++t) acc[t] = zero;

  async_cp16(fsrc + fk, fdst[0]);
  for (int k0 = 0; k0 < 1024; k0 += 32) {
    const int buf = (k0 >> 5) & 1;
    const bool more = (k0 + 32) < 1024;
    if (more) {
      async_cp16(fsrc + k0 + 32 + fk, fdst[buf ^ 1]);
      WAIT_ASYNC(1);
    } else {
      WAIT_ASYNC(0);
    }
    __syncthreads();
    const v16h a = cat8(*(const v8h*)(arow + k0 + half * 16),
                        *(const v8h*)(arow + k0 + half * 16 + 8));
#pragma unroll
    for (int t = 0; t < 4; ++t) {
      const v16h bf = *(const v16h*)&wtile[buf][t * 16 + r][half * 16];
      acc[t] = wmma16(a, bf, acc[t]);
    }
    __syncthreads();
  }
#pragma unroll
  for (int t = 0; t < 4; ++t)
#pragma unroll
    for (int v = 0; v < 8; ++v) {
      const size_t idx = (size_t)(n0 + v + 8 * half) * 1024 + f0 + t * 16 + r;
      out[idx] = x[idx] + acc[t][v];
    }
}

// ---------------------------------------------------------------------------
extern "C" void kernel_launch(void* const* d_in, const int* in_sizes, int n_in,
                              void* d_out, int out_size, void* d_ws, size_t ws_size,
                              hipStream_t stream) {
  const float* x     = (const float*)d_in[0];
  const float* Wq    = (const float*)d_in[1];
  const float* Wkv   = (const float*)d_in[2];
  const float* Wo    = (const float*)d_in[3];
  const float* ln_g  = (const float*)d_in[4];
  const float* ln_b  = (const float*)d_in[5];
  const float* lnc_g = (const float*)d_in[6];
  const float* lnc_b = (const float*)d_in[7];
  float* out = (float*)d_out;

  char* ws = (char*)d_ws;
  size_t off = 0;
  auto take = [&](size_t bytes) -> char* {
    char* p = ws + off;
    off += (bytes + 255) & ~(size_t)255;
    return p;
  };
  _Float16* xn   = (_Float16*)take((size_t)8192 * 1024 * 2);   // 16 MB
  _Float16* cnb  = (_Float16*)take((size_t)8192 * 1024 * 2);   // 16 MB
  _Float16* wq_h = (_Float16*)take((size_t)1024 * 1024 * 2);   //  2 MB
  _Float16* wkv_h= (_Float16*)take((size_t)2048 * 1024 * 2);   //  4 MB
  _Float16* wo_h = (_Float16*)take((size_t)1024 * 1024 * 2);   //  2 MB
  _Float16* Qh   = (_Float16*)take((size_t)64 * 2048 * 64 * 2);// 16 MB
  _Float16* Kh   = (_Float16*)take((size_t)64 * 2048 * 64 * 2);// 16 MB
  _Float16* Vt   = (_Float16*)take((size_t)64 * 64 * 2048 * 2);// 16 MB
  _Float16* Oh   = (_Float16*)take((size_t)8192 * 1024 * 2);   // 16 MB
  (void)ws_size; (void)in_sizes; (void)n_in; (void)out_size;

  k_ln2_cast<<<8192, 256, 0, stream>>>(x, ln_g, ln_b, lnc_g, lnc_b, xn, cnb);
  k_cvt_f16<<<1024, 256, 0, stream>>>(Wq,  wq_h,  1024 * 1024);
  k_cvt_f16<<<1024, 256, 0, stream>>>(Wkv, wkv_h, 2048 * 1024);
  k_cvt_f16<<<1024, 256, 0, stream>>>(Wo,  wo_h,  1024 * 1024);
  k_gemm_qkv<<<3072, 256, 0, stream>>>(xn, cnb, wq_h, wkv_h, Qh, Kh, Vt);
  k_attn<<<1024, 256, 0, stream>>>(Qh, Kh, Vt, Oh);
  k_gemm_out<<<1024, 256, 0, stream>>>(Oh, wo_h, x, out);
}
